// MultiHeadAttention_8358006358291
// MI455X (gfx1250) — matrix operand order repaired and checked by reading, not compiled, not run on hardware
//
#include <hip/hip_runtime.h>
#include <hip/hip_bf16.h>

// ---------------- problem constants ----------------
#define DMODEL 1024
#define NHEADS 16
#define HDIM   64
#define SEQ    2048
#define BATCH  4
#define STOT   (BATCH * SEQ)   // 8192 flattened (b,s) rows

// ---------------- WMMA types ----------------
typedef __attribute__((ext_vector_type(16))) _Float16 v16h;
typedef __attribute__((ext_vector_type(8)))  _Float16 v8h;
typedef __attribute__((ext_vector_type(8)))  float    v8f;

#define WMMA_F16(a, b, c) \
  __builtin_amdgcn_wmma_f32_16x16x32_f16(false, (a), false, (b), (short)0, (c), false, false)

static __device__ inline v8h ld8(const _Float16* p) { return *(const v8h*)p; }

// A fragment: 16x32 f16 tile from row-major storage (ld in halfs).
// ISA layout (16-bit A 16x32): lane L -> row M=L%16, half=L/16:
//   elems 0..7  -> K = half*8 + i ; elems 8..15 -> K = 16 + half*8 + (i-8)
static __device__ inline v16h load_a_frag(const _Float16* base, int ld) {
  int lane = threadIdx.x & 31;
  int row  = lane & 15;
  int half = lane >> 4;
  v8h lo = ld8(base + (size_t)row * ld + half * 8);
  v8h hi = ld8(base + (size_t)row * ld + 16 + half * 8);
  v16h a;
#pragma unroll
  for (int i = 0; i < 8; i++) { a[i] = lo[i]; a[8 + i] = hi[i]; }
  return a;
}

// B fragment: 32x16 f16 tile from B-transposed (n-major) storage bt[n][k], ld in halfs.
// ISA layout (16-bit B 32x16): lane -> col N=lane%16, kh=lane/16; elems 0..7 -> K = kh*8 + i ; elems 8..15 -> K = 16 + kh*8 + (i-8).
static __device__ inline v16h load_b_frag(const _Float16* bt, int ld) {
  int lane = threadIdx.x & 31;
  int n  = lane & 15;
  int kh = lane >> 4;
  const _Float16* p = bt + (size_t)n * ld + kh * 8;
  v8h lo = ld8(p);
  v8h hi = ld8(p + 16);
  v16h b;
#pragma unroll
  for (int i = 0; i < 8; i++) { b[i] = lo[i]; b[8 + i] = hi[i]; }
  return b;
}

// ---------------- kernel 0: fp32 -> fp16 ----------------
__global__ __launch_bounds__(256) void cvt_f32_to_f16(const float* __restrict__ src,
                                                      _Float16* __restrict__ dst, int n) {
  int i = blockIdx.x * blockDim.x + threadIdx.x;
  int stride = gridDim.x * blockDim.x;
  for (; i < n; i += stride) dst[i] = (_Float16)src[i];
}

// ---------------- kernel 1: per-head QKV projection ----------------
// grid: (STOT/128, NHEADS, 3[q,k,v]); block 128 (4 waves).
// Wave tile: 32 rows x 64 cols. K-loop unrolled 2x with ping-pong fragment
// sets A/B so the prefetch needs no register rotation (no v_mov churn).
__global__ __launch_bounds__(128) void qkv_proj(
    const _Float16* __restrict__ x16,
    const _Float16* __restrict__ wq, const _Float16* __restrict__ wk, const _Float16* __restrict__ wv,
    const float* __restrict__ bq, const float* __restrict__ bk, const float* __restrict__ bv,
    _Float16* __restrict__ q, _Float16* __restrict__ k, _Float16* __restrict__ v) {
  int h     = blockIdx.y;
  int which = blockIdx.z;
  const _Float16* W    = (which == 0) ? wq : (which == 1) ? wk : wv;
  const float*    bias = (which == 0) ? bq : (which == 1) ? bk : bv;
  _Float16*       out  = (which == 0) ? q  : (which == 1) ? k  : v;

  int wave = threadIdx.x >> 5;
  int lane = threadIdx.x & 31;
  int row0 = blockIdx.x * 128 + wave * 32;               // flattened (b,s) row
  const _Float16* Wh   = W + (size_t)h * HDIM * DMODEL;
  const _Float16* xrow = x16 + (size_t)row0 * DMODEL;

  v8f acc[2][4] = {};

  // set A: fragments for k = 0
  v16h a0A = load_a_frag(xrow, DMODEL);
  v16h a1A = load_a_frag(xrow + (size_t)16 * DMODEL, DMODEL);
  v16h bA[4];
#pragma unroll
  for (int nt = 0; nt < 4; nt++) bA[nt] = load_b_frag(Wh + (size_t)(nt * 16) * DMODEL, DMODEL);

  for (int k0 = 0; k0 < DMODEL; k0 += 64) {
    // set B: fragments for k0+32 (always in-bounds; DMODEL % 64 == 0)
    v16h a0B = load_a_frag(xrow + k0 + 32, DMODEL);
    v16h a1B = load_a_frag(xrow + (size_t)16 * DMODEL + k0 + 32, DMODEL);
    v16h bB[4];
#pragma unroll
    for (int nt = 0; nt < 4; nt++)
      bB[nt] = load_b_frag(Wh + (size_t)(nt * 16) * DMODEL + k0 + 32, DMODEL);

    // compute on set A (k0)
#pragma unroll
    for (int nt = 0; nt < 4; nt++) {
      acc[0][nt] = WMMA_F16(a0A, bA[nt], acc[0][nt]);
      acc[1][nt] = WMMA_F16(a1A, bA[nt], acc[1][nt]);
    }

    // set A: fragments for k0+64 (wrap to 0 on last iter; harmless reload)
    int kn = (k0 + 64 < DMODEL) ? (k0 + 64) : 0;
    a0A = load_a_frag(xrow + kn, DMODEL);
    a1A = load_a_frag(xrow + (size_t)16 * DMODEL + kn, DMODEL);
#pragma unroll
    for (int nt = 0; nt < 4; nt++)
      bA[nt] = load_b_frag(Wh + (size_t)(nt * 16) * DMODEL + kn, DMODEL);

    // compute on set B (k0+32)
#pragma unroll
    for (int nt = 0; nt < 4; nt++) {
      acc[0][nt] = WMMA_F16(a0B, bB[nt], acc[0][nt]);
      acc[1][nt] = WMMA_F16(a1B, bB[nt], acc[1][nt]);
    }
  }

  int n = lane & 15, half = lane >> 4;
  int bidx = row0 / SEQ;               // 128-row blocks never straddle a batch
  int s    = row0 % SEQ;
  size_t obase = (size_t)(bidx * NHEADS + h) * SEQ + s;
#pragma unroll
  for (int rb = 0; rb < 2; rb++) {
#pragma unroll
    for (int nt = 0; nt < 4; nt++) {
      float bb = bias[h * HDIM + nt * 16 + n];
#pragma unroll
      for (int r = 0; r < 8; r++) {
        int M = rb * 16 + r + 8 * half;
        out[(obase + M) * HDIM + nt * 16 + n] = (_Float16)(acc[rb][nt][r] + bb);
      }
    }
  }
}

// ---------------- kernel 2: flash attention ----------------
// grid: (SEQ/128, NHEADS, BATCH); block 256 (8 waves, 16 query rows each).
// K/V tiles double-buffered in LDS: next tile prefetched into registers during
// compute, one barrier per 32-key step.
__global__ __launch_bounds__(256) void flash_attn(
    const _Float16* __restrict__ q, const _Float16* __restrict__ k,
    const _Float16* __restrict__ v, _Float16* __restrict__ z16) {
  __shared__ alignas(16) _Float16 Klds[2][32][HDIM];     // K tile [t][e], 2 x 4KB
  __shared__ alignas(16) _Float16 Vt[2][HDIM][32];       // V tile transposed [e][t], 2 x 4KB
  __shared__ alignas(16) _Float16 Plds[8][16][32];       // per-wave P staging, 8KB

  int qb = blockIdx.x, h = blockIdx.y, b = blockIdx.z;
  int wave = threadIdx.x >> 5, lane = threadIdx.x & 31;
  size_t head_base = (size_t)(b * NHEADS + h) * SEQ * HDIM;
  int s0 = qb * 128 + wave * 16;

  // cooperative-load indexing: 256 threads x 8 halfs = 32x64 tile
  int tid  = threadIdx.x;
  int trow = tid >> 3;              // 0..31
  int col0 = (tid & 7) * 8;         // 0..56

  // Q fragments, pre-scaled by 1/sqrt(HDIM)
  v16h aq0 = load_a_frag(q + head_base + (size_t)s0 * HDIM, HDIM);
  v16h aq1 = load_a_frag(q + head_base + (size_t)s0 * HDIM + 32, HDIM);
  {
    const _Float16 sc = (_Float16)0.125f;
#pragma unroll
    for (int i = 0; i < 16; i++) { aq0[i] *= sc; aq1[i] *= sc; }
  }

  v8f zacc[4] = {};
  float m[8], l[8];
#pragma unroll
  for (int r = 0; r < 8; r++) { m[r] = -1e30f; l[r] = 0.0f; }

  // prologue: stage tile t0 = 0 into buffer 0
  {
    v8h kreg = ld8(k + head_base + (size_t)trow * HDIM + col0);
    v8h vreg = ld8(v + head_base + (size_t)trow * HDIM + col0);
    *(v8h*)&Klds[0][trow][col0] = kreg;
#pragma unroll
    for (int j = 0; j < 8; j++) Vt[0][col0 + j][trow] = vreg[j];
  }
  __syncthreads();

  int buf = 0;
  for (int t0 = 0; t0 < SEQ; t0 += 32) {
    // prefetch next tile into registers (redundant reload of t=0 on last iter)
    int tn = (t0 + 32 < SEQ) ? (t0 + 32) : 0;
    v8h kreg = ld8(k + head_base + (size_t)(tn + trow) * HDIM + col0);
    v8h vreg = ld8(v + head_base + (size_t)(tn + trow) * HDIM + col0);

    // ---- scores: two 16x16 tiles over this 32-key block ----
    v8f st[2];
#pragma unroll
    for (int j = 0; j < 2; j++) {
      v8f c = {};
      v16h b0 = load_b_frag(&Klds[buf][16 * j][0], HDIM);       // e 0..31
      v16h b1 = load_b_frag(&Klds[buf][16 * j][0] + 32, HDIM);  // e 32..63
      c = WMMA_F16(aq0, b0, c);
      c = WMMA_F16(aq1, b1, c);
      st[j] = c;
    }

    // ---- online softmax per row slot (row M = r + 8*(lane/16) across 16 lanes) ----
#pragma unroll
    for (int r = 0; r < 8; r++) {
      float x0 = st[0][r];
      float x1 = st[1][r];
      float mt = fmaxf(x0, x1);
#pragma unroll
      for (int off = 1; off < 16; off <<= 1) mt = fmaxf(mt, __shfl_xor(mt, off, 32));
      float mnew = fmaxf(m[r], mt);
      float p0 = __expf(x0 - mnew);
      float p1 = __expf(x1 - mnew);
      float rs = p0 + p1;
#pragma unroll
      for (int off = 1; off < 16; off <<= 1) rs += __shfl_xor(rs, off, 32);
      float alpha = __expf(m[r] - mnew);
      l[r] = l[r] * alpha + rs;
      m[r] = mnew;
#pragma unroll
      for (int nt = 0; nt < 4; nt++) zacc[nt][r] *= alpha;
      st[0][r] = p0;
      st[1][r] = p1;
    }

    // ---- stage P (C-layout -> row-major [16][32] f16), reload as A fragment ----
    {
      int n = lane & 15, half = lane >> 4;
#pragma unroll
      for (int r = 0; r < 8; r++) {
        int M = r + 8 * half;
        Plds[wave][M][n]      = (_Float16)st[0][r];
        Plds[wave][M][16 + n] = (_Float16)st[1][r];
      }
    }
    v16h ap = load_a_frag(&Plds[wave][0][0], 32);  // same-wave LDS ops are in-order

    // ---- z += P[16x32] x V[32x64] ----
#pragma unroll
    for (int nt = 0; nt < 4; nt++) {
      v16h bf = load_b_frag(&Vt[buf][16 * nt][0], 32);
      zacc[nt] = WMMA_F16(ap, bf, zacc[nt]);
    }

    // ---- write prefetched tile into the other buffer ----
    int nb = buf ^ 1;
    *(v8h*)&Klds[nb][trow][col0] = kreg;
#pragma unroll
    for (int j = 0; j < 8; j++) Vt[nb][col0 + j][trow] = vreg[j];
    __syncthreads();
    buf = nb;
  }

  // ---- normalize and write concat-head layout ----
  {
    int n = lane & 15, half = lane >> 4;
#pragma unroll
    for (int nt = 0; nt < 4; nt++) {
#pragma unroll
      for (int r = 0; r < 8; r++) {
        int M = r + 8 * half;
        float val = zacc[nt][r] / l[r];
        z16[((size_t)(b * SEQ + s0 + M)) * DMODEL + h * HDIM + nt * 16 + n] = (_Float16)val;
      }
    }
  }
}

// ---------------- kernel 3: output projection ----------------
// grid: (STOT/128, DMODEL/64); block 128 (4 waves). Wave tile 32x64,
// ping-pong pipelined K-loop (no register rotation).
__global__ __launch_bounds__(128) void out_proj(
    const _Float16* __restrict__ z16, const _Float16* __restrict__ wo,
    const float* __restrict__ bo, float* __restrict__ out) {
  int wave = threadIdx.x >> 5, lane = threadIdx.x & 31;
  int row0 = blockIdx.x * 128 + wave * 32;
  int col0 = blockIdx.y * 64;
  const _Float16* zrow = z16 + (size_t)row0 * DMODEL;
  const _Float16* wcol = wo + (size_t)col0 * DMODEL;

  v8f acc[2][4] = {};

  v16h a0A = load_a_frag(zrow, DMODEL);
  v16h a1A = load_a_frag(zrow + (size_t)16 * DMODEL, DMODEL);
  v16h bA[4];
#pragma unroll
  for (int nt = 0; nt < 4; nt++) bA[nt] = load_b_frag(wcol + (size_t)(nt * 16) * DMODEL, DMODEL);

  for (int k0 = 0; k0 < DMODEL; k0 += 64) {
    v16h a0B = load_a_frag(zrow + k0 + 32, DMODEL);
    v16h a1B = load_a_frag(zrow + (size_t)16 * DMODEL + k0 + 32, DMODEL);
    v16h bB[4];
#pragma unroll
    for (int nt = 0; nt < 4; nt++)
      bB[nt] = load_b_frag(wcol + (size_t)(nt * 16) * DMODEL + k0 + 32, DMODEL);

#pragma unroll
    for (int nt = 0; nt < 4; nt++) {
      acc[0][nt] = WMMA_F16(a0A, bA[nt], acc[0][nt]);
      acc[1][nt] = WMMA_F16(a1A, bA[nt], acc[1][nt]);
    }

    int kn = (k0 + 64 < DMODEL) ? (k0 + 64) : 0;
    a0A = load_a_frag(zrow + kn, DMODEL);
    a1A = load_a_frag(zrow + (size_t)16 * DMODEL + kn, DMODEL);
#pragma unroll
    for (int nt = 0; nt < 4; nt++)
      bA[nt] = load_b_frag(wcol + (size_t)(nt * 16) * DMODEL + kn, DMODEL);

#pragma unroll
    for (int nt = 0; nt < 4; nt++) {
      acc[0][nt] = WMMA_F16(a0B, bB[nt], acc[0][nt]);
      acc[1][nt] = WMMA_F16(a1B, bB[nt], acc[1][nt]);
    }
  }

  int n = lane & 15, half = lane >> 4;
#pragma unroll
  for (int rb = 0; rb < 2; rb++) {
#pragma unroll
    for (int nt = 0; nt < 4; nt++) {
      float bb = bo[col0 + nt * 16 + n];
#pragma unroll
      for (int r = 0; r < 8; r++) {
        int M = rb * 16 + r + 8 * half;
        out[(size_t)(row0 + M) * DMODEL + col0 + nt * 16 + n] = acc[rb][nt][r] + bb;
      }
    }
  }
}

// ---------------- host launch ----------------
extern "C" void kernel_launch(void* const* d_in, const int* in_sizes, int n_in,
                              void* d_out, int out_size, void* d_ws, size_t ws_size,
                              hipStream_t stream) {
  (void)in_sizes; (void)n_in; (void)out_size; (void)ws_size;
  const float* x  = (const float*)d_in[0];
  const float* Wq = (const float*)d_in[1];
  const float* bq = (const float*)d_in[2];
  const float* Wk = (const float*)d_in[3];
  const float* bk = (const float*)d_in[4];
  const float* Wv = (const float*)d_in[5];
  const float* bv = (const float*)d_in[6];
  const float* Wo = (const float*)d_in[7];
  const float* bo = (const float*)d_in[8];
  float* out = (float*)d_out;

  // workspace carve-out (fp16 buffers), 256B-aligned chunks
  char* ws = (char*)d_ws;
  size_t off = 0;
  auto carve = [&](size_t elems) -> _Float16* {
    _Float16* p = (_Float16*)(ws + off);
    off += (elems * sizeof(_Float16) + 255) & ~(size_t)255;
    return p;
  };
  const size_t HEADQKV = (size_t)BATCH * NHEADS * SEQ * HDIM;  // 8,388,608
  _Float16* x16  = carve((size_t)STOT * DMODEL);
  _Float16* wq16 = carve((size_t)NHEADS * HDIM * DMODEL);
  _Float16* wk16 = carve((size_t)NHEADS * HDIM * DMODEL);
  _Float16* wv16 = carve((size_t)NHEADS * HDIM * DMODEL);
  _Float16* wo16 = carve((size_t)DMODEL * DMODEL);
  _Float16* q16  = carve(HEADQKV);
  _Float16* k16  = carve(HEADQKV);
  _Float16* v16  = carve(HEADQKV);
  _Float16* z16  = carve((size_t)STOT * DMODEL);

  // conversions
  cvt_f32_to_f16<<<1024, 256, 0, stream>>>(x,  x16,  STOT * DMODEL);
  cvt_f32_to_f16<<<256,  256, 0, stream>>>(Wq, wq16, NHEADS * HDIM * DMODEL);
  cvt_f32_to_f16<<<256,  256, 0, stream>>>(Wk, wk16, NHEADS * HDIM * DMODEL);
  cvt_f32_to_f16<<<256,  256, 0, stream>>>(Wv, wv16, NHEADS * HDIM * DMODEL);
  cvt_f32_to_f16<<<256,  256, 0, stream>>>(Wo, wo16, DMODEL * DMODEL);

  // QKV projections
  dim3 gproj(STOT / 128, NHEADS, 3);
  qkv_proj<<<gproj, 128, 0, stream>>>(x16, wq16, wk16, wv16, bq, bk, bv, q16, k16, v16);

  // flash attention
  dim3 gattn(SEQ / 128, NHEADS, BATCH);
  flash_attn<<<gattn, 256, 0, stream>>>(q16, k16, v16, z16);

  // output projection
  dim3 gout(STOT / 128, DMODEL / 64);
  out_proj<<<gout, 128, 0, stream>>>(z16, wo16, bo, out);
}
